// Attention_18872086299409
// MI455X (gfx1250) — matrix operand order repaired and checked by reading, compiled, not run on hardware
//
#include <hip/hip_runtime.h>
#include <hip/hip_bf16.h>

// ---------------------------------------------------------------------------
// Types for CDNA5 WMMA (wave32): v16bf A/B fragments, v8f accumulator.
// ---------------------------------------------------------------------------
typedef __bf16 v16bf __attribute__((ext_vector_type(16)));
typedef float  v8f   __attribute__((ext_vector_type(8)));
typedef unsigned short ushortT;
typedef ushortT ushort8 __attribute__((ext_vector_type(8)));

struct U16x2 { ushort8 lo, hi; };

__device__ __forceinline__ v16bf load_frag(const ushortT* p0, const ushortT* p1) {
    U16x2 t;
    t.lo = *(const ushort8*)p0;
    t.hi = *(const ushort8*)p1;
    return __builtin_bit_cast(v16bf, t);
}

__device__ __forceinline__ v8f wmma_bf16(v16bf a, v16bf b, v8f c) {
    // D = A(16x32 bf16) * B(32x16 bf16) + C(16x16 f32)
    return __builtin_amdgcn_wmma_f32_16x16x32_bf16(
        /*neg_a=*/false, a, /*neg_b=*/false, b,
        /*c_mod=*/(short)0, c, /*reuse_a=*/false, /*reuse_b=*/false);
}

__device__ __forceinline__ ushortT f2bf(float f) {
    unsigned int u = __builtin_bit_cast(unsigned int, f);
    unsigned int r = u + 0x7FFFu + ((u >> 16) & 1u);   // round-to-nearest-even
    return (ushortT)(r >> 16);
}

// ---------------------------------------------------------------------------
// CDNA5 async global -> LDS copy (16 B per lane), tracked by ASYNCcnt.
// ---------------------------------------------------------------------------
__device__ __forceinline__ void async_copy16(const void* gptr, void* lptr) {
    unsigned lds = (unsigned)(uintptr_t)lptr;   // low 32 bits = LDS offset
    asm volatile("global_load_async_to_lds_b128 %0, %1, off"
                 :: "v"(lds), "v"(gptr) : "memory");
}

__device__ __forceinline__ void wait_async0() {
#if __has_builtin(__builtin_amdgcn_s_wait_asynccnt)
    __builtin_amdgcn_s_wait_asynccnt(0);
#else
    asm volatile("s_wait_asynccnt 0x0" ::: "memory");
#endif
}

// ---------------------------------------------------------------------------
// Problem constants
// ---------------------------------------------------------------------------
#define DIMV   512
#define HEADS  8
#define DHEAD  64
#define BATCH  4
#define SEQ    2048
#define ROWS   (BATCH * SEQ)        // 8192
#define SCALE  0.04419417382415922f // 512^-0.5 (reference uses full dim)

// workspace layout (bytes)
#define OFF_XB     ((size_t)0)                    //  8 MiB  x as bf16 [8192,512]
#define OFF_WQKVT  ((size_t)8388608)              //  1.5MiB w_qkv^T bf16 [1536,512]
#define OFF_WOUTT  ((size_t)9961472)              //  0.5MiB w_out^T bf16 [512,512]
#define OFF_QB     ((size_t)10485760)             //  8 MiB  Q bf16 [b*h,2048,64]
#define OFF_KB     ((size_t)18874368)             //  8 MiB  K bf16 [b*h,2048,64]
#define OFF_VTB    ((size_t)27262976)             //  8 MiB  V^T bf16 [b*h,64,2048]
#define OFF_ATTNB  ((size_t)35651584)             //  8 MiB  attn out bf16 [8192,512]

#define LDB_LDS 136    // 128 K-chunk + 8 pad (272 B row ≡ 4 banks mod 64)

// ---------------------------------------------------------------------------
// 16(M) x 64(N) tile GEMM with block-shared, double-buffered async-LDS B panel.
// A: bf16 row-major (per-wave 16 rows, global).  BT: bf16 [64, 512] panel.
// Each of the 8 waves accumulates its own 16x64 tile; all share the B panel.
// ---------------------------------------------------------------------------
__device__ __forceinline__ void stage_b_chunk(ushortT* sBb,
                                              const ushortT* __restrict__ BT,
                                              int kc, int tid) {
    // 64 rows x 128 elems = 1024 x 16B chunks; 256 threads x 4 copies
#pragma unroll
    for (int i = 0; i < 4; ++i) {
        const int idx = tid + i * 256;
        const int row = idx >> 4;            // 0..63
        const int off = (idx & 15) * 8;      // 0..120 elems
        async_copy16(BT + (size_t)row * DIMV + kc + off,
                     sBb + (size_t)row * LDB_LDS + off);
    }
}

__device__ __forceinline__ void gemm16x64_lds(const ushortT* __restrict__ A, int lda,
                                              const ushortT* __restrict__ BT,
                                              v8f acc[4]) {
    __shared__ __align__(16) ushortT sB[2][64 * LDB_LDS];

    const int tid  = threadIdx.x;
    const int lane = tid & 31;
    const int m    = lane & 15;
    const int half = lane >> 4;
    const ushortT* Ar = A + (size_t)m * lda;

    stage_b_chunk(sB[0], BT, 0, tid);
    wait_async0();
    __syncthreads();

    int cur = 0;
    for (int kc = 0; kc < DIMV; kc += 128) {
        const int nxt = cur ^ 1;
        if (kc + 128 < DIMV) {
            stage_b_chunk(sB[nxt], BT, kc + 128, tid);       // overlap DMA
            __builtin_prefetch(Ar + kc + 128, 0, 3);         // WGP-scope prefetch
        }
#pragma unroll
        for (int k0 = 0; k0 < 128; k0 += 32) {
            // A 16x32 fragment: lane holds K = half*8..+8 and 16+half*8..+8
            const ushortT* ap = Ar + kc + k0 + half * 8;
            v16bf a = load_frag(ap, ap + 16);
            // hoist all 4 B fragments before the WMMAs -> staggered DScnt waits
            v16bf b[4];
#pragma unroll
            for (int t = 0; t < 4; ++t) {
                const ushortT* bp = &sB[cur][(size_t)(t * 16 + m) * LDB_LDS
                                             + k0 + half * 8];
                b[t] = load_frag(bp, bp + 16);
            }
#pragma unroll
            for (int t = 0; t < 4; ++t)
                acc[t] = wmma_bf16(a, b[t], acc[t]);
        }
        wait_async0();
        __syncthreads();
        cur = nxt;
    }
}

// ---------------------------------------------------------------------------
// Prep: fp32 -> bf16 (elementwise) and fp32 -> bf16 transposed
// ---------------------------------------------------------------------------
__global__ void cvt_bf16_kernel(const float* __restrict__ src,
                                ushortT* __restrict__ dst, int n) {
    int i = blockIdx.x * blockDim.x + threadIdx.x;
    if (i < n) dst[i] = f2bf(src[i]);
}

__global__ void transpose_bf16_kernel(const float* __restrict__ src,
                                      ushortT* __restrict__ dst, int R, int C) {
    int i = blockIdx.x * blockDim.x + threadIdx.x;
    if (i < R * C) {
        int r = i / C, c = i % C;
        dst[(size_t)c * R + r] = f2bf(src[i]);
    }
}

// ---------------------------------------------------------------------------
// Stage 1: QKV GEMM  [8192,512] x [512,1536] -> scatter into Q, K, V^T (bf16)
// Block = 8 waves = 128 rows x 64 cols; B panel shared via async LDS.
// ---------------------------------------------------------------------------
__global__ __launch_bounds__(256) void qkv_gemm_kernel(
        const ushortT* __restrict__ xb, const ushortT* __restrict__ wqkvT,
        ushortT* __restrict__ qb, ushortT* __restrict__ kb,
        ushortT* __restrict__ vtb) {
    const int wave = threadIdx.x >> 5;
    const int lane = threadIdx.x & 31;
    const int tile = blockIdx.x * 8 + wave;
    const int tm = tile & 511;          // 512 row tiles of 16 (block-contiguous)
    const int tn = tile >> 9;           // 24 col tiles of 64 (uniform per block)
    const int row0 = tm * 16, col0 = tn * 64;

    v8f acc[4] = {};
    gemm16x64_lds(xb + (size_t)row0 * DIMV, DIMV,
                  wqkvT + (size_t)col0 * DIMV, acc);

    const int m = lane & 15, half = lane >> 4;
#pragma unroll
    for (int t = 0; t < 4; ++t) {
#pragma unroll
        for (int r = 0; r < 8; ++r) {
            const int col = col0 + t * 16 + m;       // 0..1535
            const int row = row0 + half * 8 + r;     // 0..8191
            const int c  = col >> 9;                 // 0=q 1=k 2=v
            const int hh = (col >> 6) & 7;
            const int dd = col & 63;
            const int b  = row >> 11;
            const int n  = row & 2047;
            const int bh = b * HEADS + hh;
            const ushortT bv = f2bf(acc[t][r]);
            if (c == 0)      qb [((size_t)bh * SEQ + n) * DHEAD + dd] = bv;
            else if (c == 1) kb [((size_t)bh * SEQ + n) * DHEAD + dd] = bv;
            else             vtb[((size_t)bh * DHEAD + dd) * SEQ + n] = bv;
        }
    }
}

// ---------------------------------------------------------------------------
// Stage 2: flash attention. One (b,h,128-query block) per workgroup; waves own
// a 16-query tile each. K/V^T 32-key blocks are staged into LDS with async
// copies, double-buffered so the DMA of block j+32 overlaps compute of block j.
// ---------------------------------------------------------------------------
__device__ __forceinline__ void stage_kv(ushortT* sKb, ushortT* sVb,
                                         const ushortT* __restrict__ K,
                                         const ushortT* __restrict__ VT,
                                         int j0, int tid) {
    {   // K rows j0..j0+31, 64 elems each -> sK[row*72 + off]
        const int row = tid >> 3;          // 0..31
        const int off = (tid & 7) * 8;     // 0..56 elems
        async_copy16(K + (size_t)(j0 + row) * DHEAD + off, sKb + row * 72 + off);
    }
    {   // V^T rows d=0..63, cols j0..j0+31 -> sV[d*40 + off]
        const int d   = tid >> 2;          // 0..63
        const int off = (tid & 3) * 8;     // 0..24 elems
        async_copy16(VT + (size_t)d * SEQ + j0 + off, sVb + d * 40 + off);
    }
}

__global__ __launch_bounds__(256) void attn_kernel(
        const ushortT* __restrict__ qb, const ushortT* __restrict__ kb,
        const ushortT* __restrict__ vtb, ushortT* __restrict__ attnb) {
    __shared__ __align__(16) ushortT sK[2][32 * 72];   // pad: 144 B row
    __shared__ __align__(16) ushortT sV[2][64 * 40];   // pad: 80 B row
    __shared__ __align__(16) ushortT sP[8 * 16 * 40];  // per-wave P tile

    const int tid  = threadIdx.x;
    const int wave = tid >> 5;
    const int lane = tid & 31;
    const int m    = lane & 15;
    const int half = lane >> 4;

    const int qblk = blockIdx.x & 15;
    const int bh   = blockIdx.x >> 4;            // b*8 + h
    const int q0   = qblk * 128 + wave * 16;

    const ushortT* Q  = qb  + ((size_t)bh * SEQ + q0) * DHEAD;
    const ushortT* K  = kb  + (size_t)bh * SEQ * DHEAD;
    const ushortT* VT = vtb + (size_t)bh * DHEAD * SEQ;

    // preload Q fragments: two k-steps of 32 over DHEAD=64
    v16bf qf[2];
#pragma unroll
    for (int ks = 0; ks < 2; ++ks) {
        const ushortT* qr = Q + (size_t)m * DHEAD + ks * 32 + half * 8;
        qf[ks] = load_frag(qr, qr + 16);
    }

    float mrow[8], lrow[8];
    v8f O[4] = {};
#pragma unroll
    for (int r = 0; r < 8; ++r) { mrow[r] = -3.0e38f; lrow[r] = 0.0f; }

    ushortT* sp = sP + wave * 640;

    stage_kv(sK[0], sV[0], K, VT, 0, tid);
    wait_async0();
    __syncthreads();

    int cur = 0;
    for (int j0 = 0; j0 < SEQ; j0 += 32) {
        const int nxt = cur ^ 1;
        if (j0 + 32 < SEQ) stage_kv(sK[nxt], sV[nxt], K, VT, j0 + 32, tid);

        // ----- S = Q * K^T : two 16x16 tiles (j-halves), K-dim = 64 -----
        // hoist all 4 K fragments before the WMMAs -> staggered DScnt waits
        v16bf kf[4];
#pragma unroll
        for (int ks = 0; ks < 2; ++ks) {
            const ushortT* kr0 = &sK[cur][(size_t)m        * 72 + ks * 32 + half * 8];
            const ushortT* kr1 = &sK[cur][(size_t)(16 + m) * 72 + ks * 32 + half * 8];
            kf[2 * ks]     = load_frag(kr0, kr0 + 16);
            kf[2 * ks + 1] = load_frag(kr1, kr1 + 16);
        }
        v8f S0 = {}, S1 = {};
        S0 = wmma_bf16(qf[0], kf[0], S0);
        S1 = wmma_bf16(qf[0], kf[1], S1);
        S0 = wmma_bf16(qf[1], kf[2], S0);
        S1 = wmma_bf16(qf[1], kf[3], S1);

        // ----- online softmax update (fp32), row = half*8 + r -----
        float alpha[8];
#pragma unroll
        for (int r = 0; r < 8; ++r) {
            float s0 = S0[r] * SCALE;
            float s1 = S1[r] * SCALE;
            float mx = fmaxf(s0, s1);
#pragma unroll
            for (int off = 8; off >= 1; off >>= 1)
                mx = fmaxf(mx, __shfl_xor(mx, off, 32));   // within 16-lane half
            float nm = fmaxf(mrow[r], mx);
            float al = __expf(mrow[r] - nm);
            mrow[r]  = nm;
            float p0 = __expf(s0 - nm);
            float p1 = __expf(s1 - nm);
            float sm = p0 + p1;
#pragma unroll
            for (int off = 8; off >= 1; off >>= 1)
                sm += __shfl_xor(sm, off, 32);
            lrow[r] = lrow[r] * al + sm;
            alpha[r] = al;
            const int prow = half * 8 + r;
            sp[prow * 40 + m]      = f2bf(p0);
            sp[prow * 40 + 16 + m] = f2bf(p1);
        }
        // ----- reload P in A-fragment layout (16x32) from LDS -----
        const ushortT* pr = sp + m * 40 + half * 8;
        v16bf pf = load_frag(pr, pr + 16);
        // hoist all 4 V fragments before the WMMAs
        v16bf vf[4];
#pragma unroll
        for (int t = 0; t < 4; ++t) {
            const ushortT* vr = &sV[cur][(size_t)(t * 16 + m) * 40 + half * 8];
            vf[t] = load_frag(vr, vr + 16);
        }
        // ----- O = diag(alpha)*O + P * V -----
#pragma unroll
        for (int t = 0; t < 4; ++t) {
#pragma unroll
            for (int r = 0; r < 8; ++r) O[t][r] *= alpha[r];
            O[t] = wmma_bf16(pf, vf[t], O[t]);
        }

        wait_async0();      // next-block DMA finished (overlapped with compute)
        __syncthreads();    // all waves done reading cur before it is refilled
        cur = nxt;
    }

    // normalize and store to attn buffer [8192, 512] bf16, col = h*64 + d
    const int b = bh >> 3, h = bh & 7;
#pragma unroll
    for (int r = 0; r < 8; ++r) lrow[r] = 1.0f / lrow[r];
#pragma unroll
    for (int t = 0; t < 4; ++t) {
#pragma unroll
        for (int r = 0; r < 8; ++r) {
            const int row = b * SEQ + q0 + half * 8 + r;
            attnb[(size_t)row * DIMV + h * DHEAD + t * 16 + m] =
                f2bf(O[t][r] * lrow[r]);
        }
    }
}

// ---------------------------------------------------------------------------
// Stage 3: out = attn[8192,512] x w_out[512,512] + b_out  -> fp32 d_out
// ---------------------------------------------------------------------------
__global__ __launch_bounds__(256) void out_gemm_kernel(
        const ushortT* __restrict__ attnb, const ushortT* __restrict__ woutT,
        const float* __restrict__ bias, float* __restrict__ out) {
    const int wave = threadIdx.x >> 5;
    const int lane = threadIdx.x & 31;
    const int tile = blockIdx.x * 8 + wave;     // 4096 tiles
    const int tm = tile & 511;
    const int tn = tile >> 9;                   // 8 col tiles (uniform per block)
    const int row0 = tm * 16, col0 = tn * 64;

    v8f acc[4] = {};
    gemm16x64_lds(attnb + (size_t)row0 * DIMV, DIMV,
                  woutT + (size_t)col0 * DIMV, acc);

    const int m = lane & 15, half = lane >> 4;
#pragma unroll
    for (int t = 0; t < 4; ++t) {
#pragma unroll
        for (int r = 0; r < 8; ++r) {
            const int col = col0 + t * 16 + m;
            const int row = row0 + half * 8 + r;
            out[(size_t)row * DIMV + col] = acc[t][r] + bias[col];
        }
    }
}

// ---------------------------------------------------------------------------
extern "C" void kernel_launch(void* const* d_in, const int* in_sizes, int n_in,
                              void* d_out, int out_size, void* d_ws, size_t ws_size,
                              hipStream_t stream) {
    (void)in_sizes; (void)n_in; (void)out_size; (void)ws_size;
    const float* x      = (const float*)d_in[0];
    const float* w_qkv  = (const float*)d_in[1];
    const float* w_out  = (const float*)d_in[2];
    const float* b_out  = (const float*)d_in[3];
    float* out = (float*)d_out;

    char* ws = (char*)d_ws;
    ushortT* xb     = (ushortT*)(ws + OFF_XB);
    ushortT* wqkvT  = (ushortT*)(ws + OFF_WQKVT);
    ushortT* woutT  = (ushortT*)(ws + OFF_WOUTT);
    ushortT* qb     = (ushortT*)(ws + OFF_QB);
    ushortT* kb     = (ushortT*)(ws + OFF_KB);
    ushortT* vtb    = (ushortT*)(ws + OFF_VTB);
    ushortT* attnb  = (ushortT*)(ws + OFF_ATTNB);

    // prep: bf16 conversions / transposes
    cvt_bf16_kernel<<<(ROWS * DIMV + 255) / 256, 256, 0, stream>>>(x, xb, ROWS * DIMV);
    transpose_bf16_kernel<<<(DIMV * 3 * DIMV + 255) / 256, 256, 0, stream>>>(
        w_qkv, wqkvT, DIMV, 3 * DIMV);
    transpose_bf16_kernel<<<(DIMV * DIMV + 255) / 256, 256, 0, stream>>>(
        w_out, woutT, DIMV, DIMV);

    // stage 1: QKV GEMM (512 * 24 wave tiles / 8 waves per block)
    qkv_gemm_kernel<<<1536, 256, 0, stream>>>(xb, wqkvT, qb, kb, vtb);

    // stage 2: flash attention (4*8 heads * 16 query blocks)
    attn_kernel<<<512, 256, 0, stream>>>(qb, kb, vtb, attnb);

    // stage 3: output projection (512 * 8 wave tiles / 8 waves per block)
    out_gemm_kernel<<<512, 256, 0, stream>>>(attnb, woutT, b_out, out);
}
